// _PointnetSAModuleBase_40810779246761
// MI455X (gfx1250) — matrix operand order repaired and checked by reading, not compiled, not run on hardware
//
#include <hip/hip_runtime.h>
#include <hip/hip_bf16.h>

#define B_      8
#define N_      8192
#define C_      64
#define NPOINT_ 2048
#define NSAMPLE_ 32
#define RADIUS2 0.04f

typedef __attribute__((ext_vector_type(16))) _Float16 v16h;
typedef __attribute__((ext_vector_type(8)))  _Float16 v8h;
typedef __attribute__((ext_vector_type(8)))  float    v8f;

// ---------------------------------------------------------------------------
// Kernel 1: Farthest Point Sampling. One 1024-thread block per batch.
// xyz staged in dynamic LDS (96KB), min-dist in registers (8 pts/thread).
// Writes new_xyz directly (fps indices are not needed downstream).
// ---------------------------------------------------------------------------
__global__ __launch_bounds__(1024) void fps_kernel(const float* __restrict__ xyz,
                                                   float* __restrict__ new_xyz)
{
    extern __shared__ float smem[];
    float* sx = smem;
    float* sy = smem + N_;
    float* sz = smem + 2 * N_;
    __shared__ float rv[32];
    __shared__ int   ri[32];
    __shared__ int   sel;

    const int b    = blockIdx.x;
    const int t    = threadIdx.x;
    const int lane = t & 31;
    const int w    = t >> 5;
    const float* base = xyz + (size_t)b * N_ * 3;

    for (int n = t; n < N_; n += 1024) {
        sx[n] = base[n * 3 + 0];
        sy[n] = base[n * 3 + 1];
        sz[n] = base[n * 3 + 2];
    }
    __syncthreads();

    // each thread owns points n = t + p*1024
    float px[8], py[8], pz[8], md[8];
#pragma unroll
    for (int p = 0; p < 8; ++p) {
        int n = t + p * 1024;
        px[p] = sx[n]; py[p] = sy[n]; pz[p] = sz[n];
        md[p] = 3.402823e38f;
    }

    int last = 0;
    if (t == 0) {
        float* o = new_xyz + (size_t)b * NPOINT_ * 3;
        o[0] = sx[0]; o[1] = sy[0]; o[2] = sz[0];
    }

    for (int it = 1; it < NPOINT_; ++it) {
        float lx = sx[last], ly = sy[last], lz = sz[last];
        float bv = -1.0f;
        int   bi = 0x7fffffff;
#pragma unroll
        for (int p = 0; p < 8; ++p) {
            float dx = px[p] - lx, dy = py[p] - ly, dz = pz[p] - lz;
            float d  = dx * dx + dy * dy + dz * dz;
            md[p] = fminf(md[p], d);
            if (md[p] > bv) { bv = md[p]; bi = t + p * 1024; }
        }
        // wave32 reduction (argmax, lowest index on tie)
#pragma unroll
        for (int off = 16; off > 0; off >>= 1) {
            float ov = __shfl_down(bv, off, 32);
            int   oi = __shfl_down(bi, off, 32);
            if (ov > bv || (ov == bv && oi < bi)) { bv = ov; bi = oi; }
        }
        if (lane == 0) { rv[w] = bv; ri[w] = bi; }
        __syncthreads();
        if (w == 0) {
            bv = rv[lane]; bi = ri[lane];
#pragma unroll
            for (int off = 16; off > 0; off >>= 1) {
                float ov = __shfl_down(bv, off, 32);
                int   oi = __shfl_down(bi, off, 32);
                if (ov > bv || (ov == bv && oi < bi)) { bv = ov; bi = oi; }
            }
            if (lane == 0) sel = bi;
        }
        __syncthreads();
        last = sel;
        if (t == 0) {
            float* o = new_xyz + ((size_t)b * NPOINT_ + it) * 3;
            o[0] = sx[last]; o[1] = sy[last]; o[2] = sz[last];
        }
    }
}

// ---------------------------------------------------------------------------
// Kernel 2: Ball query. One thread per query point; xyz tiled through LDS.
// Takes the first NSAMPLE indices (ascending) with d2 < r^2; pads with first.
// ---------------------------------------------------------------------------
#define BQ_THREADS 256
#define BQ_TILE    1024
__global__ __launch_bounds__(BQ_THREADS) void ballq_kernel(const float* __restrict__ xyz,
                                                           const float* __restrict__ new_xyz,
                                                           int* __restrict__ idx)
{
    __shared__ float tx[BQ_TILE], ty[BQ_TILE], tz[BQ_TILE];
    const int sq = blockIdx.x * BQ_THREADS + threadIdx.x;   // global query id
    const int b  = sq / NPOINT_;                            // block never straddles batch
    const float* base = xyz + (size_t)b * N_ * 3;
    const float qx = new_xyz[(size_t)sq * 3 + 0];
    const float qy = new_xyz[(size_t)sq * 3 + 1];
    const float qz = new_xyz[(size_t)sq * 3 + 2];

    int cnt = 0, first = 0;
    int* out = idx + (size_t)sq * NSAMPLE_;

    for (int t0 = 0; t0 < N_; t0 += BQ_TILE) {
        __syncthreads();
        for (int j = threadIdx.x; j < BQ_TILE; j += BQ_THREADS) {
            const float* p = base + (size_t)(t0 + j) * 3;
            tx[j] = p[0]; ty[j] = p[1]; tz[j] = p[2];
        }
        __syncthreads();
        if (cnt < NSAMPLE_) {
            for (int j = 0; j < BQ_TILE; ++j) {
                float dx = tx[j] - qx, dy = ty[j] - qy, dz = tz[j] - qz;
                float d2 = dx * dx + dy * dy + dz * dz;
                if (d2 < RADIUS2) {
                    if (cnt == 0) first = t0 + j;
                    out[cnt] = t0 + j;
                    if (++cnt >= NSAMPLE_) break;
                }
            }
        }
    }
    for (int k = cnt; k < NSAMPLE_; ++k) out[k] = first;
}

// ---------------------------------------------------------------------------
// Kernel 3: gather + 3-layer MLP (WMMA f16 -> f32 accum) + max over samples.
// One wave per query point, 2 waves / 64-thread block. Weights (transposed,
// f16) shared in LDS; activations per-wave in LDS. A-fragments are loaded
// once per stage and reused across all output tiles (fully unrolled).
//
// Fragment layouts follow CDNA5 ISA 7.12.2:
//  A 16x32 f16 : lane<16 row=lane,  elems 0..7 = K(k0+hi*8..), 8..15 = K(k0+16+hi*8..)
//  B 32x16 f16 : lane<16 col=lane,  elems 0..7 = K(k0+hi*8..), 8..15 = K(k0+16+hi*8..)   (W stored [n][k])
//  C/D 16x16 f32: col = lane&15, row = 8*hi + v
// ---------------------------------------------------------------------------
#define MLP_WAVES 2

__device__ __forceinline__ v16h load_a_frag(const _Float16* base, int ld, int r0, int k0, int lane)
{
    const int r  = lane & 15;
    const int hi = lane >> 4;
    const _Float16* row = base + (size_t)(r0 + r) * ld + k0;
    v8h lo = *(const v8h*)(row + hi * 8);
    v8h hv = *(const v8h*)(row + 16 + hi * 8);
    v16h out;
#pragma unroll
    for (int e = 0; e < 8; ++e) { out[e] = lo[e]; out[8 + e] = hv[e]; }
    return out;
}

__device__ __forceinline__ v16h load_b_frag(const _Float16* wt, int ld, int n0, int k0, int lane)
{
    const int c  = lane & 15;
    const int hi = lane >> 4;
    const _Float16* row = wt + (size_t)(n0 + c) * ld + k0;
    v8h lo = *(const v8h*)(row + hi * 8);
    v8h hv = *(const v8h*)(row + 16 + hi * 8);
    v16h out;
#pragma unroll
    for (int e = 0; e < 8; ++e) { out[e] = lo[e]; out[8 + e] = hv[e]; }
    return out;
}

__global__ __launch_bounds__(MLP_WAVES * 32) void mlp_kernel(
    const float* __restrict__ xyz, const float* __restrict__ features,
    const float* __restrict__ new_xyz, const int* __restrict__ idx,
    const float* __restrict__ W1, const float* __restrict__ b1,
    const float* __restrict__ W2, const float* __restrict__ b2,
    const float* __restrict__ W3, const float* __restrict__ b3,
    float* __restrict__ new_feat)
{
    __shared__ __align__(16) _Float16 sW1t[64][96];    // [n][k], K padded 67->96
    __shared__ __align__(16) _Float16 sW2t[64][64];
    __shared__ __align__(16) _Float16 sW3t[128][64];
    __shared__ float sb1[64], sb2[64], sb3[128];
    __shared__ __align__(16) _Float16 sG[MLP_WAVES][32][96];  // g, reused for h2
    __shared__ __align__(16) _Float16 sH[MLP_WAVES][32][64];  // h1

    const int tid  = threadIdx.x;
    const int w    = tid >> 5;
    const int lane = tid & 31;

    // ---- cooperative weight load (f32 -> f16, transposed, padded) ----
    for (int i = tid; i < 64 * 96; i += MLP_WAVES * 32) {
        int n = i / 96, k = i % 96;
        sW1t[n][k] = (k < 67) ? (_Float16)W1[k * 64 + n] : (_Float16)0.0f;
    }
    for (int i = tid; i < 64 * 64; i += MLP_WAVES * 32) {
        int n = i >> 6, k = i & 63;
        sW2t[n][k] = (_Float16)W2[k * 64 + n];
    }
    for (int i = tid; i < 128 * 64; i += MLP_WAVES * 32) {
        int n = i >> 6, k = i & 63;
        sW3t[n][k] = (_Float16)W3[k * 128 + n];
    }
    for (int i = tid; i < 64; i += MLP_WAVES * 32) { sb1[i] = b1[i]; sb2[i] = b2[i]; }
    for (int i = tid; i < 128; i += MLP_WAVES * 32) sb3[i] = b3[i];

    // ---- gather g: 32 samples x (3 centered xyz + 64 feats + 29 zero pad) ----
    const int sq = blockIdx.x * MLP_WAVES + w;      // query id in [0, B*NPOINT)
    const int b  = sq / NPOINT_;
    const int s  = sq % NPOINT_;
    {
        const int pid = idx[(size_t)sq * NSAMPLE_ + lane];     // lane = sample row
        const float* ctr = new_xyz + (size_t)sq * 3;
        const float* p   = xyz + ((size_t)b * N_ + pid) * 3;
        _Float16* grow = &sG[w][lane][0];
        grow[0] = (_Float16)(p[0] - ctr[0]);
        grow[1] = (_Float16)(p[1] - ctr[1]);
        grow[2] = (_Float16)(p[2] - ctr[2]);
        const float* f = features + (size_t)b * C_ * N_ + pid;
        for (int c = 0; c < 64; ++c) grow[3 + c] = (_Float16)f[(size_t)c * N_];
#pragma unroll
        for (int c = 67; c < 96; ++c) grow[c] = (_Float16)0.0f;
    }
    __syncthreads();

    const int c  = lane & 15;
    const int hi = lane >> 4;

    // ---- stage 1: h1 = relu(g @ W1 + b1)   M=32 N=64 K=96 ----
    {
        v16h a0[3], a1[3];
#pragma unroll
        for (int kc = 0; kc < 3; ++kc) {
            a0[kc] = load_a_frag(&sG[w][0][0], 96,  0, kc * 32, lane);
            a1[kc] = load_a_frag(&sG[w][0][0], 96, 16, kc * 32, lane);
        }
#pragma unroll
        for (int nt = 0; nt < 4; ++nt) {
            v8f acc0 = {}, acc1 = {};
#pragma unroll
            for (int kc = 0; kc < 3; ++kc) {
                v16h bf = load_b_frag(&sW1t[0][0], 96, nt * 16, kc * 32, lane);
                acc0 = __builtin_amdgcn_wmma_f32_16x16x32_f16(false, a0[kc], false, bf, (short)0, acc0, false, false);
                acc1 = __builtin_amdgcn_wmma_f32_16x16x32_f16(false, a1[kc], false, bf, (short)0, acc1, false, false);
            }
            float bv = sb1[nt * 16 + c];
#pragma unroll
            for (int v = 0; v < 8; ++v) {
                float x0 = acc0[v] + bv; x0 = x0 > 0.0f ? x0 : 0.0f;
                float x1 = acc1[v] + bv; x1 = x1 > 0.0f ? x1 : 0.0f;
                sH[w][hi * 8 + v][nt * 16 + c]      = (_Float16)x0;
                sH[w][16 + hi * 8 + v][nt * 16 + c] = (_Float16)x1;
            }
        }
    }

    // ---- stage 2: h2 = relu(h1 @ W2 + b2)  M=32 N=64 K=64, h2 -> sG ----
    {
        v16h a0[2], a1[2];
#pragma unroll
        for (int kc = 0; kc < 2; ++kc) {
            a0[kc] = load_a_frag(&sH[w][0][0], 64,  0, kc * 32, lane);
            a1[kc] = load_a_frag(&sH[w][0][0], 64, 16, kc * 32, lane);
        }
#pragma unroll
        for (int nt = 0; nt < 4; ++nt) {
            v8f acc0 = {}, acc1 = {};
#pragma unroll
            for (int kc = 0; kc < 2; ++kc) {
                v16h bf = load_b_frag(&sW2t[0][0], 64, nt * 16, kc * 32, lane);
                acc0 = __builtin_amdgcn_wmma_f32_16x16x32_f16(false, a0[kc], false, bf, (short)0, acc0, false, false);
                acc1 = __builtin_amdgcn_wmma_f32_16x16x32_f16(false, a1[kc], false, bf, (short)0, acc1, false, false);
            }
            float bv = sb2[nt * 16 + c];
#pragma unroll
            for (int v = 0; v < 8; ++v) {
                float x0 = acc0[v] + bv; x0 = x0 > 0.0f ? x0 : 0.0f;
                float x1 = acc1[v] + bv; x1 = x1 > 0.0f ? x1 : 0.0f;
                sG[w][hi * 8 + v][nt * 16 + c]      = (_Float16)x0;
                sG[w][16 + hi * 8 + v][nt * 16 + c] = (_Float16)x1;
            }
        }
    }

    // ---- stage 3: h3 = relu(h2 @ W3 + b3)  M=32 N=128 K=64, fused max over M ----
    {
        v16h a0[2], a1[2];
#pragma unroll
        for (int kc = 0; kc < 2; ++kc) {
            a0[kc] = load_a_frag(&sG[w][0][0], 96,  0, kc * 32, lane);
            a1[kc] = load_a_frag(&sG[w][0][0], 96, 16, kc * 32, lane);
        }
#pragma unroll
        for (int nt = 0; nt < 8; ++nt) {
            v8f acc0 = {}, acc1 = {};
#pragma unroll
            for (int kc = 0; kc < 2; ++kc) {
                v16h bf = load_b_frag(&sW3t[0][0], 64, nt * 16, kc * 32, lane);
                acc0 = __builtin_amdgcn_wmma_f32_16x16x32_f16(false, a0[kc], false, bf, (short)0, acc0, false, false);
                acc1 = __builtin_amdgcn_wmma_f32_16x16x32_f16(false, a1[kc], false, bf, (short)0, acc1, false, false);
            }
            float bv = sb3[nt * 16 + c];
            float pm = 0.0f;                                  // relu floor
#pragma unroll
            for (int v = 0; v < 8; ++v) {
                pm = fmaxf(pm, acc0[v] + bv);
                pm = fmaxf(pm, acc1[v] + bv);
            }
            // lanes c and c+16 hold complementary rows of column c
            pm = fmaxf(pm, __shfl_xor(pm, 16, 32));
            if (lane < 16)
                new_feat[(size_t)b * 128 * NPOINT_ + (size_t)(nt * 16 + c) * NPOINT_ + s] = pm;
        }
    }
}

// ---------------------------------------------------------------------------
extern "C" void kernel_launch(void* const* d_in, const int* in_sizes, int n_in,
                              void* d_out, int out_size, void* d_ws, size_t ws_size,
                              hipStream_t stream)
{
    const float* xyz      = (const float*)d_in[0];
    const float* features = (const float*)d_in[1];
    const float* W1       = (const float*)d_in[2];
    const float* b1       = (const float*)d_in[3];
    const float* W2       = (const float*)d_in[4];
    const float* b2       = (const float*)d_in[5];
    const float* W3       = (const float*)d_in[6];
    const float* b3       = (const float*)d_in[7];

    float* out      = (float*)d_out;
    float* new_xyz  = out;                                 // (B, NPOINT, 3)
    float* new_feat = out + (size_t)B_ * NPOINT_ * 3;      // (B, 128, NPOINT)
    int*   idx      = (int*)d_ws;                          // (B*NPOINT, NSAMPLE)

    fps_kernel<<<B_, 1024, 3 * N_ * sizeof(float), stream>>>(xyz, new_xyz);
    ballq_kernel<<<(B_ * NPOINT_) / BQ_THREADS, BQ_THREADS, 0, stream>>>(xyz, new_xyz, idx);
    mlp_kernel<<<(B_ * NPOINT_) / MLP_WAVES, MLP_WAVES * 32, 0, stream>>>(
        xyz, features, new_xyz, idx, W1, b1, W2, b2, W3, b3, new_feat);
}
